// _GCNPatchBlock_43181601194898
// MI455X (gfx1250) — compile-verified, matrix operand order checked by reading, not run on hardware
//
#include <hip/hip_runtime.h>

// ---------------- problem constants ----------------
#define Bc 16
#define Nc 32
#define Pc 128
#define Dc 256
#define Ec 1024
#define Hc 8
#define Lc 2
#define DFc 1024
#define Gc (Bc * Nc)          // 512 graphs
#define Mc (Gc * Pc)          // 65536 rows
#define NEc (Ec + Pc)         // 1152 edges incl self-loops

// ---------------- WMMA types / helpers ----------------
typedef __attribute__((ext_vector_type(16))) __bf16 v16bf;
typedef __attribute__((ext_vector_type(8)))  float  v8f;

__device__ __forceinline__ v8f wmma_bf16(v16bf a, v16bf b, v8f c) {
  return __builtin_amdgcn_wmma_f32_16x16x32_bf16(
      /*neg_a=*/false, a, /*neg_b=*/false, b,
      /*c_mod=*/(short)0, c, /*reuse_a=*/false, /*reuse_b=*/false);
}

// Load one 16x32 bf16 fragment (A layout; B uses same layout on [N][K] storage).
// lane<16: rows M=lane, K = {0..7, 16..23}; lane>=16: M=lane-16, K = {8..15, 24..31}
__device__ __forceinline__ v16bf load_frag(const __bf16* __restrict__ base,
                                           int ld, int lane) {
  const int m  = lane & 15;
  const int kb = (lane >> 4) << 3;            // 0 or 8
  const __bf16* p0 = base + m * ld + kb;      // K = kb .. kb+7
  const __bf16* p1 = base + m * ld + 16 + kb; // K = 16+kb .. +7
  v16bf f;
#pragma unroll
  for (int i = 0; i < 8; ++i) f[i] = p0[i];
#pragma unroll
  for (int i = 0; i < 8; ++i) f[8 + i] = p1[i];
  return f;
}

template <typename T> __device__ __forceinline__ __bf16 as_bf16(T v);
template <> __device__ __forceinline__ __bf16 as_bf16<float>(float v)   { return (__bf16)v; }
template <> __device__ __forceinline__ __bf16 as_bf16<__bf16>(__bf16 v) { return v; }

template <typename T> __device__ __forceinline__ T from_f32(float v);
template <> __device__ __forceinline__ float  from_f32<float>(float v)  { return v; }
template <> __device__ __forceinline__ __bf16 from_f32<__bf16>(float v) { return (__bf16)v; }

// ---------------- generic GEMM: C = act(A[MxK] @ W[KxN] + bias) ----------------
// 256 threads (8 waves); block tile 128x128; wave tile 32x64 (2x4 WMMA, 8/step)
template <typename TIN, typename TOUT, bool RELU>
__global__ __launch_bounds__(256) void gemm_t(
    const TIN* __restrict__ A, const float* __restrict__ W,
    const float* __restrict__ bias, TOUT* __restrict__ C,
    int M, int N, int K) {
  extern __shared__ char smem[];
  __bf16* a_s = (__bf16*)smem;                        // [128][32]
  __bf16* w_s = (__bf16*)(smem + 128 * 32 * 2);       // [128(N)][32(K)] (transposed)
  const int tid  = threadIdx.x;
  const int lane = tid & 31;
  const int wave = tid >> 5;
  const long bm = (long)blockIdx.x * 128;
  const long bn = (long)blockIdx.y * 128;
  const int r0 = (wave >> 1) * 32;                    // 0,32,64,96
  const int c0 = (wave & 1) * 64;                     // 0,64

  v8f acc[2][4] = {};
  const int nk = K >> 5;
  for (int kt = 0; kt < nk; ++kt) {
    if (kt + 1 < nk) {                                // hint next K-tile into cache
      __builtin_prefetch(&A[(bm + (tid >> 5)) * (long)K + (kt + 1) * 32 + (tid & 31)], 0, 1);
      __builtin_prefetch(&W[((kt + 1) * 32 + (tid >> 7)) * (long)N + bn + (tid & 127)], 0, 1);
    }
    __syncthreads();
#pragma unroll
    for (int i = 0; i < 16; ++i) {                    // A tile: 128x32
      int idx = tid + i * 256;
      int row = idx >> 5, col = idx & 31;
      a_s[idx] = as_bf16<TIN>(A[(bm + row) * (long)K + kt * 32 + col]);
    }
#pragma unroll
    for (int i = 0; i < 16; ++i) {                    // W tile: 32x128 -> [n][k]
      int idx = tid + i * 256;
      int kk = idx >> 7, nn = idx & 127;
      w_s[nn * 32 + kk] = (__bf16)W[(kt * 32 + kk) * (long)N + bn + nn];
    }
    __syncthreads();
    v16bf a0 = load_frag(a_s + r0 * 32, 32, lane);
    v16bf a1 = load_frag(a_s + (r0 + 16) * 32, 32, lane);
    v16bf b[4];
#pragma unroll
    for (int j = 0; j < 4; ++j)
      b[j] = load_frag(w_s + (c0 + j * 16) * 32, 32, lane);
#pragma unroll
    for (int j = 0; j < 4; ++j) {
      acc[0][j] = wmma_bf16(a0, b[j], acc[0][j]);
      acc[1][j] = wmma_bf16(a1, b[j], acc[1][j]);
    }
  }
  const int n_in  = lane & 15;
  const int rbase = (lane >> 4) * 8;
#pragma unroll
  for (int i = 0; i < 2; ++i) {
#pragma unroll
    for (int j = 0; j < 4; ++j) {
      long gcol = bn + c0 + j * 16 + n_in;
      float bv = bias ? bias[gcol] : 0.0f;
#pragma unroll
      for (int r = 0; r < 8; ++r) {
        long grow = bm + r0 + i * 16 + rbase + r;
        float v = acc[i][j][r] + bv;
        if (RELU) v = fmaxf(v, 0.0f);
        C[grow * (long)N + gcol] = from_f32<TOUT>(v);
      }
    }
  }
}

// ---------------- shared-graph normalization ----------------
__global__ __launch_bounds__(256) void norm_prep_kernel(
    const int* __restrict__ EI, float* __restrict__ norm) {
  __shared__ float deg[Pc];
  __shared__ float dinv[Pc];
  int tid = threadIdx.x;
  if (tid < Pc) deg[tid] = 0.0f;
  __syncthreads();
  for (int e = tid; e < NEc; e += 256) {
    int d = (e < Ec) ? EI[Ec + e] : (e - Ec);
    atomicAdd(&deg[d], 1.0f);
  }
  __syncthreads();
  if (tid < Pc) dinv[tid] = deg[tid] > 0.0f ? rsqrtf(deg[tid]) : 0.0f;
  __syncthreads();
  for (int e = tid; e < NEc; e += 256) {
    int s = (e < Ec) ? EI[e] : (e - Ec);
    int d = (e < Ec) ? EI[Ec + e] : (e - Ec);
    norm[e] = dinv[s] * dinv[d];
  }
}

// ---------------- GCN scatter-aggregate (one WG per graph, big-LDS) ----------------
__global__ __launch_bounds__(256) void gcn_aggregate_kernel(
    const float* __restrict__ XW, const int* __restrict__ EI,
    const float* __restrict__ norm, const float* __restrict__ bias,
    float* __restrict__ Hout) {
  extern __shared__ char smem[];
  float* xw_s  = (float*)smem;                       // 128x256 f32 (128 KB)
  float* acc_s = xw_s + Pc * Dc;                     // 128x256 f32 (128 KB)
  int*   src_s = (int*)(acc_s + Pc * Dc);
  int*   dst_s = src_s + NEc;
  float* nrm_s = (float*)(dst_s + NEc);

  const int g   = blockIdx.x;
  const int tid = threadIdx.x;
  const long gb = (long)g * Pc * Dc;
  float4* xw4        = (float4*)xw_s;
  float4* acc4       = (float4*)acc_s;
  const float4* gxw4 = (const float4*)(XW + gb);
  for (int i = tid; i < Pc * Dc / 4; i += 256) {
    xw4[i]  = gxw4[i];
    acc4[i] = make_float4(0.f, 0.f, 0.f, 0.f);
  }
  for (int e = tid; e < NEc; e += 256) {
    int s, d;
    if (e < Ec) { s = EI[e]; d = EI[Ec + e]; }
    else        { s = e - Ec; d = s; }
    src_s[e] = s; dst_s[e] = d; nrm_s[e] = norm[e];
  }
  __syncthreads();
  for (int t = tid; t < NEc * (Dc / 4); t += 256) {
    int e = t >> 6, c = t & 63;
    int s = src_s[e], d = dst_s[e];
    float w = nrm_s[e];
    float4 m = xw4[s * (Dc / 4) + c];
    float* dp = acc_s + d * Dc + c * 4;
    atomicAdd(dp + 0, m.x * w);
    atomicAdd(dp + 1, m.y * w);
    atomicAdd(dp + 2, m.z * w);
    atomicAdd(dp + 3, m.w * w);
  }
  __syncthreads();
  for (int i = tid; i < Pc * Dc; i += 256) {
    int d = i & (Dc - 1);
    Hout[gb + i] = fmaxf(acc_s[i] + bias[d], 0.0f);
  }
}

// ---------------- per-(graph,head) attention (bf16 in / bf16 out) ----------------
#define HD 32
__global__ __launch_bounds__(256) void attention_kernel(
    const __bf16* __restrict__ Q, const __bf16* __restrict__ Km,
    const __bf16* __restrict__ V, __bf16* __restrict__ O) {
  extern __shared__ char smem[];
  __bf16* q_s  = (__bf16*)smem;                 // [128][32]
  __bf16* k_s  = q_s + Pc * HD;                 // [128][32]  (== B [n][k] for QK^T)
  __bf16* vt_s = k_s + Pc * HD;                 // [32][128]  (V^T: [n][k] for PV)
  float*  sc_s = (float*)(vt_s + HD * Pc);      // [128][128] f32 scores
  __bf16* pr_s = (__bf16*)(sc_s + Pc * Pc);     // [128][128] bf16 probs

  const int g = blockIdx.x >> 3;
  const int h = blockIdx.x & 7;
  const int tid = threadIdx.x, lane = tid & 31, wave = tid >> 5;
  const long base = (long)g * Pc * Dc + h * HD;

  for (int i = tid; i < Pc * HD; i += 256) {
    int p = i >> 5, d = i & 31;
    long gi = base + (long)p * Dc + d;
    q_s[p * HD + d]  = Q[gi];
    k_s[p * HD + d]  = Km[gi];
    vt_s[d * Pc + p] = V[gi];
  }
  __syncthreads();

  const float scale = 0.17677669529663687f;     // 1/sqrt(32)
  {
    v16bf a = load_frag(q_s + wave * 16 * HD, HD, lane);
    const int n_in = lane & 15, rbase = (lane >> 4) * 8;
#pragma unroll
    for (int nt = 0; nt < 8; ++nt) {
      v16bf b = load_frag(k_s + nt * 16 * HD, HD, lane);
      v8f acc = {};
      acc = wmma_bf16(a, b, acc);
#pragma unroll
      for (int r = 0; r < 8; ++r)
        sc_s[(wave * 16 + rbase + r) * Pc + nt * 16 + n_in] = acc[r] * scale;
    }
  }
  __syncthreads();

  if (tid < Pc) {                               // one softmax row per thread
    float* row = sc_s + tid * Pc;
    float m = -3.4e38f;
    for (int j = 0; j < Pc; ++j) m = fmaxf(m, row[j]);
    float s = 0.0f;
    for (int j = 0; j < Pc; ++j) { float e = __expf(row[j] - m); row[j] = e; s += e; }
    float inv = 1.0f / s;
    for (int j = 0; j < Pc; ++j) pr_s[tid * Pc + j] = (__bf16)(row[j] * inv);
  }
  __syncthreads();

  {
    v8f acc0 = {}, acc1 = {};
#pragma unroll
    for (int kt = 0; kt < 4; ++kt) {
      v16bf a  = load_frag(pr_s + wave * 16 * Pc + kt * 32, Pc, lane);
      v16bf b0 = load_frag(vt_s + 0  * Pc + kt * 32, Pc, lane);
      v16bf b1 = load_frag(vt_s + 16 * Pc + kt * 32, Pc, lane);
      acc0 = wmma_bf16(a, b0, acc0);
      acc1 = wmma_bf16(a, b1, acc1);
    }
    const int n_in = lane & 15, rbase = (lane >> 4) * 8;
#pragma unroll
    for (int r = 0; r < 8; ++r) {
      long row = wave * 16 + rbase + r;
      O[base + row * Dc + n_in]      = (__bf16)acc0[r];
      O[base + row * Dc + 16 + n_in] = (__bf16)acc1[r];
    }
  }
}

// ------- Out = LayerNorm(X + Y) * g + b, wave-per-row (wave32); Y is f32 or bf16 -------
template <typename YT>
__global__ __launch_bounds__(256) void residual_ln_kernel(
    const float* __restrict__ X, const YT* __restrict__ Y,
    const float* __restrict__ Gm, const float* __restrict__ Bt,
    float* __restrict__ Out, int Mrows) {
  int wave = threadIdx.x >> 5, lane = threadIdx.x & 31;
  long row = (long)blockIdx.x * 8 + wave;
  if (row >= Mrows) return;
  const float* x = X + row * Dc;
  const YT*    y = Y + row * Dc;
  float v[8];
  float s = 0.0f;
#pragma unroll
  for (int i = 0; i < 8; ++i) {
    v[i] = x[lane + 32 * i] + (float)y[lane + 32 * i];
    s += v[i];
  }
#pragma unroll
  for (int off = 16; off; off >>= 1) s += __shfl_xor(s, off, 32);
  float mu = s * (1.0f / Dc);
  float vs = 0.0f;
#pragma unroll
  for (int i = 0; i < 8; ++i) { float d = v[i] - mu; vs += d * d; }
#pragma unroll
  for (int off = 16; off; off >>= 1) vs += __shfl_xor(vs, off, 32);
  float rs = rsqrtf(vs * (1.0f / Dc) + 1e-5f);
#pragma unroll
  for (int i = 0; i < 8; ++i) {
    int c = lane + 32 * i;
    Out[row * Dc + c] = (v[i] - mu) * rs * Gm[c] + Bt[c];
  }
}

// ---------------- host orchestration ----------------
extern "C" void kernel_launch(void* const* d_in, const int* in_sizes, int n_in,
                              void* d_out, int out_size, void* d_ws, size_t ws_size,
                              hipStream_t stream) {
  const float* x       = (const float*)d_in[0];
  const int*   ei      = (const int*)d_in[1];
  const float* conv1_W = (const float*)d_in[6];
  const float* conv1_b = (const float*)d_in[7];
  const float* conv2_W = (const float*)d_in[8];
  const float* conv2_b = (const float*)d_in[9];
  const float* Wq = (const float*)d_in[10]; const float* bq = (const float*)d_in[11];
  const float* Wk = (const float*)d_in[12]; const float* bk = (const float*)d_in[13];
  const float* Wv = (const float*)d_in[14]; const float* bv = (const float*)d_in[15];
  const float* Wo = (const float*)d_in[16]; const float* bo = (const float*)d_in[17];
  const float* W1 = (const float*)d_in[18]; const float* b1 = (const float*)d_in[19];
  const float* W2 = (const float*)d_in[20]; const float* b2 = (const float*)d_in[21];
  const float* ln1_g = (const float*)d_in[22]; const float* ln1_b = (const float*)d_in[23];
  const float* ln2_g = (const float*)d_in[24]; const float* ln2_b = (const float*)d_in[25];

  float* h    = (float*)d_out;                      // persistent activations [M x D] f32
  float* wsf  = (float*)d_ws;
  float* norm = wsf;                                // NEc floats
  const size_t BUF = (size_t)Mc * Dc;               // capacity: Mc*Dc floats each
  float* buf1 = wsf + 2048;
  float* buf2 = buf1 + BUF;
  float* buf3 = buf2 + BUF;
  float* buf4 = buf3 + BUF;
  __bf16* q_b   = (__bf16*)buf1;                    // bf16 views (half occupancy)
  __bf16* k_b   = (__bf16*)buf2;
  __bf16* v_b   = (__bf16*)buf3;
  __bf16* o_b   = (__bf16*)buf4;
  __bf16* mid_b = (__bf16*)buf2;                    // FFN intermediate (bf16)
  __bf16* f2_b  = (__bf16*)buf3;                    // FFN output (bf16)
  float*  proj  = buf1;                             // attn projection (f32)

  const size_t gemm_lds = 2 * 128 * 32 * sizeof(__bf16);                      // 16 KB
  const size_t agg_lds  = 2 * (size_t)Pc * Dc * sizeof(float)
                        + NEc * (2 * sizeof(int) + sizeof(float));            // ~270 KB
  const size_t attn_lds = (size_t)3 * Pc * HD * sizeof(__bf16)
                        + (size_t)Pc * Pc * sizeof(float)
                        + (size_t)Pc * Pc * sizeof(__bf16);                   // 120 KB

  dim3 gD(Mc / 128, Dc / 128);                      // 65536x256 GEMMs

  // ---- graph normalization ----
  norm_prep_kernel<<<1, 256, 0, stream>>>(ei, norm);

  // ---- GCN conv1 / conv2 (linear -> scatter-add -> bias -> ReLU) ----
  gemm_t<float, float, false><<<gD, 256, gemm_lds, stream>>>(x, conv1_W, nullptr, buf1, Mc, Dc, Dc);
  gcn_aggregate_kernel<<<Gc, 256, agg_lds, stream>>>(buf1, ei, norm, conv1_b, h);
  gemm_t<float, float, false><<<gD, 256, gemm_lds, stream>>>(h, conv2_W, nullptr, buf1, Mc, Dc, Dc);
  gcn_aggregate_kernel<<<Gc, 256, agg_lds, stream>>>(buf1, ei, norm, conv2_b, h);

  // ---- transformer layers ----
  for (int l = 0; l < Lc; ++l) {
    const size_t wo = (size_t)l * Dc * Dc;
    gemm_t<float, __bf16, false><<<gD, 256, gemm_lds, stream>>>(h, Wq + wo, bq + l * Dc, q_b, Mc, Dc, Dc);
    gemm_t<float, __bf16, false><<<gD, 256, gemm_lds, stream>>>(h, Wk + wo, bk + l * Dc, k_b, Mc, Dc, Dc);
    gemm_t<float, __bf16, false><<<gD, 256, gemm_lds, stream>>>(h, Wv + wo, bv + l * Dc, v_b, Mc, Dc, Dc);
    attention_kernel<<<Gc * Hc, 256, attn_lds, stream>>>(q_b, k_b, v_b, o_b);
    gemm_t<__bf16, float, false><<<gD, 256, gemm_lds, stream>>>(o_b, Wo + wo, bo + l * Dc, proj, Mc, Dc, Dc);
    residual_ln_kernel<float><<<Mc / 8, 256, 0, stream>>>(h, proj, ln1_g + l * Dc, ln1_b + l * Dc, h, Mc);

    // FFN, row-chunked so the 1024-wide bf16 intermediate fits one buffer
    const int CH = 16384;
    for (int c = 0; c < Mc; c += CH) {
      gemm_t<float, __bf16, true><<<dim3(CH / 128, DFc / 128), 256, gemm_lds, stream>>>(
          h + (size_t)c * Dc, W1 + (size_t)l * Dc * DFc, b1 + l * DFc,
          mid_b, CH, DFc, Dc);
      gemm_t<__bf16, __bf16, false><<<dim3(CH / 128, Dc / 128), 256, gemm_lds, stream>>>(
          mid_b, W2 + (size_t)l * DFc * Dc, b2 + l * Dc,
          f2_b + (size_t)c * Dc, CH, Dc, DFc);
    }
    residual_ln_kernel<__bf16><<<Mc / 8, 256, 0, stream>>>(h, f2_b, ln2_g + l * Dc, ln2_b + l * Dc, h, Mc);
  }
}
